// SlidingWindowProcessor_10204842295631
// MI455X (gfx1250) — matrix operand order repaired and checked by reading, not compiled, not run on hardware
//
#include <hip/hip_runtime.h>
#include <hip/hip_bf16.h>

typedef __attribute__((ext_vector_type(16))) __bf16 v16bf;
typedef __attribute__((ext_vector_type(8)))  float  v8f;
typedef __attribute__((ext_vector_type(4)))  unsigned int v4u;
typedef __attribute__((ext_vector_type(8)))  int    v8i;
typedef __attribute__((ext_vector_type(4)))  int    v4i;

#if defined(__has_builtin)
#if __has_builtin(__builtin_amdgcn_tensor_load_to_lds) && __has_builtin(__builtin_amdgcn_s_wait_tensorcnt)
#define HAVE_TDM 1
#endif
#endif
#ifndef HAVE_TDM
#define HAVE_TDM 0
#endif

// Problem constants
#define BN   8
#define DM   256
#define LN   4096
#define HH   8
#define WN   128
#define STR  64
#define NWIN 63            // (4096-128)/64 + 1
#define EPSF 1e-5f
#define SOFTMAX_SCL 0.25504914f   // log2(e)/sqrt(32)

// LDS layout (dynamic):
//  region A: 196608 B:
//    - Phase 0/1: x window tile f32 [256][128] (TDM target, 131072 B) + LN1 reduce
//      scratch at +131072
//    - Phase 2+:  qkv bf16 [128][768]  (q-slot reused for attn output o,
//      k-slot bytes [512:1024) reused for LN2 stats, v-slot [1024:1536) for r_hi f32)
//  region B: 66048 B = max( h bf16 [128][258], per-wave S scratch 8x(16x128 f32),
//                           r_lo f32 [128][128] )
#define LDH      258
#define LDS_A    (128*768*2)
#define LDS_B    66048
#define LDS_TOT  (LDS_A + LDS_B)

// Workspace offsets (bytes)
#define WS_INSWZ   0u
#define WS_OUTSWZ  393216u            // 768*256*2
#define WS_TCNR    524288u            // +256*256*2
#define WS_S       622592u            // +32*3*256*4 ; s: 8*32*8064 f32 = 8257536

// ---------------- WMMA fragment helpers (contiguous k = 16*(lane>>4) + s in A and B, wave32) ----------------

__device__ __forceinline__ v8f wmma_bf16(v16bf a, v16bf b, v8f c) {
  return __builtin_amdgcn_wmma_f32_16x16x32_bf16(false, a, false, b, (short)0, c, false, false);
}

// A (16x32 bf16): lane holds row m=lane&15; VGPR j -> K = (lane>>4)*16 + 2j (+1)
__device__ __forceinline__ v16bf load_a_frag_lds(const __bf16* base, int ld, int row0, int k0, int lane) {
  const __bf16* p = base + (row0 + (lane & 15)) * ld + k0;
  const int kh = (lane >> 4) << 4;
  v16bf a;
#pragma unroll
  for (int j = 0; j < 8; ++j) {
    int kk = kh + (j << 1);
    a[2 * j]     = p[kk];
    a[2 * j + 1] = p[kk + 1];
  }
  return a;
}

// Same A layout, but from f32 scratch with a per-lane row scale (softmax P)
__device__ __forceinline__ v16bf load_a_frag_f32(const float* base, int ld, int k0, float scale, int lane) {
  const float* p = base + (lane & 15) * ld + k0;
  const int kh = (lane >> 4) << 4;
  v16bf a;
#pragma unroll
  for (int j = 0; j < 8; ++j) {
    int kk = kh + (j << 1);
    a[2 * j]     = (__bf16)(p[kk] * scale);
    a[2 * j + 1] = (__bf16)(p[kk + 1] * scale);
  }
  return a;
}

// B (32x16 bf16), B[k][n] stored at base[k*ld + n]: lane holds col n=lane&15,
// lane-half selects K half (0..15 / 16..31), VGPR j -> K = half*16 + 2j (+1)
__device__ __forceinline__ v16bf load_b_frag_lds(const __bf16* base, int ld, int k0, int n0, int lane) {
  const __bf16* p = base + (k0 + ((lane >> 4) << 4)) * ld + n0 + (lane & 15);
  v16bf b;
#pragma unroll
  for (int j = 0; j < 8; ++j) {
    b[2 * j]     = p[(2 * j) * ld];
    b[2 * j + 1] = p[(2 * j + 1) * ld];
  }
  return b;
}

// B where B[k][n] = src[n*ld + k] (i.e. K^T from row-major K): contiguous in k per lane
__device__ __forceinline__ v16bf load_bT_frag_lds(const __bf16* src, int ld, int n0, int lane) {
  const __bf16* p = src + (n0 + (lane & 15)) * ld + ((lane >> 4) << 4);
  v16bf b;
#pragma unroll
  for (int j = 0; j < 8; ++j) { b[2 * j] = p[2 * j]; b[2 * j + 1] = p[2 * j + 1]; }
  return b;
}

// Pre-swizzled global B frag: one contiguous 32B load per lane
__device__ __forceinline__ v16bf load_b_frag_swz(const __bf16* buf, int KC, int nt, int kc, int lane) {
  return *(const v16bf*)(buf + ((((nt * KC + kc) << 5) + lane) << 4));
}

// ---------------- Kernel 1: weight prep (bf16 swizzle + tcn repack) ----------------

__global__ void prep_weights(const float* __restrict__ inW, const float* __restrict__ outW,
                             const float* __restrict__ tcnW,
                             __bf16* __restrict__ inSwz, __bf16* __restrict__ outSwz,
                             float* __restrict__ tcnR) {
  int t = blockIdx.x * blockDim.x + threadIdx.x;
  if (t < 768 * 256) {  // in_proj: N=768, K=256, NT=48, KC=8
    int m = t & 15, l = (t >> 4) & 31, kc = (t >> 9) & 7, nt = t >> 12;
    int k = kc * 32 + ((l >> 4) << 4) + m;
    int n = nt * 16 + (l & 15);
    inSwz[t] = (__bf16)inW[n * 256 + k];
  }
  if (t < 256 * 256) {  // out_proj: N=256, K=256, NT=16, KC=8
    int m = t & 15, l = (t >> 4) & 31, kc = (t >> 9) & 7, nt = t >> 12;
    int k = kc * 32 + ((l >> 4) << 4) + m;
    int n = nt * 16 + (l & 15);
    outSwz[t] = (__bf16)outW[n * 256 + k];
  }
  if (t < 32 * 3 * 256) {  // (32,256,3) -> [c][dw][ci]
    int ci = t & 255, dw = (t >> 8) % 3, c = t / 768;
    tcnR[(c * 3 + dw) * 256 + ci] = tcnW[(c * 256 + ci) * 3 + dw];
  }
}

// ---------------- Kernel 2: fused per-window pipeline ----------------

__global__ void __launch_bounds__(256, 1)
swp_main(const float* __restrict__ x,
         const float* __restrict__ ln1w, const float* __restrict__ ln1b,
         const float* __restrict__ inprojb, const float* __restrict__ outprojb,
         const float* __restrict__ ln2w, const float* __restrict__ ln2b,
         const float* __restrict__ tcnb,
         const float* __restrict__ bng, const float* __restrict__ bnb,
         const float* __restrict__ bnm, const float* __restrict__ bnv,
         const __bf16* __restrict__ wInSwz, const __bf16* __restrict__ wOutSwz,
         const float* __restrict__ tcnR, float* __restrict__ s_out) {
  extern __shared__ char smem[];
  __bf16* qkv = (__bf16*)smem;            // region A (phase 2+)
  char*   regB = smem + LDS_A;
  __bf16* hbf = (__bf16*)regB;            // h bf16 [128][LDH]
  float*  rlo = (float*)regB;             // later: r f32 [128][128]

  const int tid  = threadIdx.x;
  const int lane = tid & 31;
  const int wv   = tid >> 5;
  const int bw   = blockIdx.x;
  const int b    = bw / NWIN, nw = bw % NWIN;
  const long xbase = (long)b * DM * LN + (long)nw * STR;

  // ---------- Phase 0: TDM DMA of the x window tile into LDS region A ----------
  // 2D tile: 256 rows (ci) x 128 f32 (w), row stride 4096 elements.
#if HAVE_TDM
  if (wv == 0) {
    unsigned long long ga = (unsigned long long)(const void*)(x + xbase);
    v4u g0;
    g0[0] = 1u;                                  // count=1 (valid user descriptor)
    g0[1] = 0u;                                  // lds_addr = 0 (region A base)
    g0[2] = (unsigned)(ga & 0xFFFFFFFFu);        // global_addr[31:0]
    g0[3] = (unsigned)((ga >> 32) & 0x01FFFFFFu) // global_addr[56:32]
            | (2u << 30);                        // type = 2 ("image")
    v8i g1;
    g1[0] = 0x00020000;                          // data_size=2 (4B); no multicast/pad/iter
    g1[1] = (int)((4096u & 0xFFFFu) << 16);      // tensor_dim0[15:0] = 4096
    g1[2] = (int)((4096u >> 16) | ((256u & 0xFFFFu) << 16)); // dim0 hi | tensor_dim1 lo (=256)
    g1[3] = (int)((128u) << 16);                 // tensor_dim1 hi=0 | tile_dim0 = 128
    g1[4] = (int)256u;                           // tile_dim1 = 256 | tile_dim2 = 0
    g1[5] = 4096;                                // tensor_dim0_stride[31:0] = 4096
    g1[6] = 0;                                   // stride hi | dim1_stride lo (unused for 2D)
    g1[7] = 0;
    v4i gz4 = {0, 0, 0, 0};                      // groups 2/3 unused (2D tensor)
    v8i gz8 = {0, 0, 0, 0, 0, 0, 0, 0};
    __builtin_amdgcn_tensor_load_to_lds(g0, g1, gz4, gz4, gz8, 0);
    __builtin_amdgcn_s_wait_tensorcnt(0);
  }
  __syncthreads();
  const float* xs = (const float*)smem;          // [256][128] in LDS
#define XVAL(ci, w) xs[(ci) * 128 + (w)]
#else
#define XVAL(ci, w) x[xbase + (long)(ci) * LN + (w)]
#endif

  // ---------- Phase 1: LN1 global stats over the whole window ----------
  float sum = 0.f, sq = 0.f;
  for (int e = tid; e < WN * DM; e += 256) {
    int ci = e >> 7, w = e & 127;
    float v = XVAL(ci, w);
    sum += v; sq += v * v;
  }
#pragma unroll
  for (int off = 16; off; off >>= 1) {
    sum += __shfl_xor(sum, off, 32);
    sq  += __shfl_xor(sq, off, 32);
  }
  float* red = (float*)(smem + 131072);  // region A tail, disjoint from x tile
  if (lane == 0) { red[wv * 2] = sum; red[wv * 2 + 1] = sq; }
  __syncthreads();
  float mu1, rstd1;
  {
    float ts = 0.f, tq = 0.f;
#pragma unroll
    for (int i = 0; i < 8; ++i) { ts += red[2 * i]; tq += red[2 * i + 1]; }
    mu1 = ts * (1.0f / (WN * DM));
    rstd1 = rsqrtf(tq * (1.0f / (WN * DM)) - mu1 * mu1 + EPSF);
  }

  // ---------- Phase 1b: h = LN1(x) in bf16 (GEMM A operand) ----------
  for (int e = tid; e < WN * DM; e += 256) {
    int ci = e >> 7, w = e & 127;
    float v = XVAL(ci, w);
    hbf[w * LDH + ci] = (__bf16)((v - mu1) * rstd1 * ln1w[w * DM + ci] + ln1b[w * DM + ci]);
  }
  __syncthreads();
#undef XVAL

  // ---------- Phase 2: QKV = h @ in_proj^T + b  (128x256 @ 256x768, WMMA) ----------
  for (int rt = 0; rt < 8; ++rt) {
    v16bf afr[8];
#pragma unroll
    for (int kc = 0; kc < 8; ++kc) afr[kc] = load_a_frag_lds(hbf, LDH, rt * 16, kc * 32, lane);
    for (int ct = wv; ct < 48; ct += 8) {
      v8f acc = {};
#pragma unroll
      for (int kc = 0; kc < 8; ++kc)
        acc = wmma_bf16(afr[kc], load_b_frag_swz(wInSwz, 8, ct, kc, lane), acc);
      int n = ct * 16 + (lane & 15);
      float bias = inprojb[n];
      int mb = rt * 16 + ((lane >> 4) << 3);
#pragma unroll
      for (int r = 0; r < 8; ++r) qkv[(mb + r) * 768 + n] = (__bf16)(acc[r] + bias);
    }
  }
  __syncthreads();

  // ---------- Phase 3: attention, one head per wave ----------
  float* Ssc = (float*)regB + wv * 2048;  // 16x128 f32 per wave
  const int head = wv;
  for (int qt = 0; qt < 8; ++qt) {
    v16bf aq = load_a_frag_lds(qkv, 768, qt * 16, head * 32, lane);
#pragma unroll
    for (int ct = 0; ct < 8; ++ct) {
      v8f z = {};
      v8f sfr = wmma_bf16(aq, load_bT_frag_lds(qkv + 256 + head * 32, 768, ct * 16, lane), z);
      int n = ct * 16 + (lane & 15);
      int mb = (lane >> 4) << 3;
#pragma unroll
      for (int r = 0; r < 8; ++r) Ssc[(mb + r) * 128 + n] = sfr[r];
    }
    __builtin_amdgcn_wave_barrier();
    // softmax: lane handles row lane&15, column half lane>>4 (64 cols)
    float invsum;
    {
      float* row = Ssc + (lane & 15) * 128 + ((lane >> 4) << 6);
      float mx = -3.4e38f;
#pragma unroll
      for (int i = 0; i < 16; ++i) {
        float4 v4 = ((const float4*)row)[i];
        mx = fmaxf(mx, fmaxf(fmaxf(v4.x, v4.y), fmaxf(v4.z, v4.w)));
      }
      mx = fmaxf(mx, __shfl_xor(mx, 16, 32));
      float ssum = 0.f;
#pragma unroll
      for (int i = 0; i < 16; ++i) {
        float4 v4 = ((float4*)row)[i];
        v4.x = exp2f((v4.x - mx) * SOFTMAX_SCL);
        v4.y = exp2f((v4.y - mx) * SOFTMAX_SCL);
        v4.z = exp2f((v4.z - mx) * SOFTMAX_SCL);
        v4.w = exp2f((v4.w - mx) * SOFTMAX_SCL);
        ((float4*)row)[i] = v4;
        ssum += v4.x + v4.y + v4.z + v4.w;
      }
      ssum += __shfl_xor(ssum, 16, 32);
      invsum = __frcp_rn(ssum);
    }
    __builtin_amdgcn_wave_barrier();
    // O = P @ V  (16x128 @ 128x32), row scale folded into A-frag conversion
#pragma unroll
    for (int nt = 0; nt < 2; ++nt) {
      v8f acc = {};
#pragma unroll
      for (int kc = 0; kc < 4; ++kc) {
        v16bf ap = load_a_frag_f32(Ssc, 128, kc * 32, invsum, lane);
        v16bf bv = load_b_frag_lds(qkv + 512 + head * 32, 768, kc * 32, nt * 16, lane);
        acc = wmma_bf16(ap, bv, acc);
      }
      int n = head * 32 + nt * 16 + (lane & 15);   // overwrite dead Q slot in place
      int mb = qt * 16 + ((lane >> 4) << 3);
#pragma unroll
      for (int r = 0; r < 8; ++r) qkv[(mb + r) * 768 + n] = (__bf16)acc[r];
    }
  }
  __syncthreads();

  // ---------- Phase 4: out_proj + exact-f32 residual, r -> LDS (split f32) ----------
  for (int rt = 0; rt < 8; ++rt) {
    v16bf afr[8];
#pragma unroll
    for (int kc = 0; kc < 8; ++kc) afr[kc] = load_a_frag_lds(qkv, 768, rt * 16, kc * 32, lane);
#pragma unroll
    for (int cti = 0; cti < 2; ++cti) {
      int ct = wv + cti * 8;
      v8f acc = {};
#pragma unroll
      for (int kc = 0; kc < 8; ++kc)
        acc = wmma_bf16(afr[kc], load_b_frag_swz(wOutSwz, 8, ct, kc, lane), acc);
      int n = ct * 16 + (lane & 15);
      float ob = outprojb[n];
      int mb = rt * 16 + ((lane >> 4) << 3);
#pragma unroll
      for (int r = 0; r < 8; ++r) {
        int m = mb + r;
        float xv = x[xbase + (long)n * LN + m];
        float h = (xv - mu1) * rstd1 * ln1w[m * DM + n] + ln1b[m * DM + n];
        float rv = acc[r] + ob + h;
        if (n < 128) rlo[m * 128 + n] = rv;
        else         ((float*)(qkv + m * 768 + 512))[n - 128] = rv;   // dead V slot
      }
    }
  }
  __syncthreads();

  // ---------- Phase 5a: LN2 row stats (2 threads / row) ----------
  {
    int m = tid >> 1, hf = tid & 1;
    float s2 = 0.f, q2 = 0.f;
    const float* p = hf ? (const float*)(qkv + m * 768 + 512) : (rlo + m * 128);
    for (int i = 0; i < 128; ++i) { float v = p[i]; s2 += v; q2 += v * v; }
    s2 += __shfl_xor(s2, 1, 32);
    q2 += __shfl_xor(q2, 1, 32);
    if (hf == 0) {
      float mu = s2 * (1.0f / 256.0f);
      float* dst = (float*)(qkv + m * 768 + 256);   // dead K slot
      dst[0] = mu;
      dst[1] = rsqrtf(q2 * (1.0f / 256.0f) - mu * mu + EPSF);
    }
  }
  __syncthreads();

  // ---------- Phase 5b: apply LN2 in place ----------
  for (int e = tid; e < WN * DM; e += 256) {
    int m = e >> 8, ci = e & 255;
    const float* st = (const float*)(qkv + m * 768 + 256);
    float mu = st[0], rs = st[1];
    float* p = (ci < 128) ? (rlo + m * 128 + ci) : ((float*)(qkv + m * 768 + 512) + (ci - 128));
    *p = (*p - mu) * rs * ln2w[ci] + ln2b[ci];
  }
  __syncthreads();

  // ---------- Phase 6: TCN conv (k=3, pad 1) + BN + ReLU + scatter into s ----------
  for (int o = tid; o < 32 * WN; o += 256) {
    int c = o >> 7, w = o & 127;
    float acc = tcnb[c];
#pragma unroll
    for (int dw = 0; dw < 3; ++dw) {
      int wp = w + dw - 1;
      if (wp < 0 || wp >= WN) continue;
      const float* wt = tcnR + (c * 3 + dw) * DM;
      const float* r0 = rlo + wp * 128;
      const float* r1 = (const float*)(qkv + wp * 768 + 512);
#pragma unroll 8
      for (int ci = 0; ci < 128; ++ci) acc += wt[ci] * r0[ci];
#pragma unroll 8
      for (int ci = 0; ci < 128; ++ci) acc += wt[128 + ci] * r1[ci];
    }
    float t = (acc - bnm[c]) * rsqrtf(bnv[c] + EPSF) * bng[c] + bnb[c];
    t = fmaxf(t, 0.0f);
    // s[b][w>>2][(w&3)*2016 + c*63 + nw]
    s_out[((long)b * 32 + (w >> 2)) * 8064 + (w & 3) * 2016 + c * 63 + nw] = t;
  }
}

// ---------------- Kernel 3: fusion conv over s (B,32,8064) ----------------

__global__ void fus_conv(const float* __restrict__ s, const float* __restrict__ fw,
                         const float* __restrict__ fb, float* __restrict__ out) {
  __shared__ float wsm[32 * 32 * 3];
  __shared__ float bsm[32];
  for (int i = threadIdx.x; i < 3072; i += blockDim.x) wsm[i] = fw[i];
  if (threadIdx.x < 32) bsm[threadIdx.x] = fb[threadIdx.x];
  __syncthreads();
  long t = (long)blockIdx.x * blockDim.x + threadIdx.x;
  const long LJ = 8064;
  if (t >= (long)BN * 32 * LJ) return;
  int j = (int)(t % LJ);
  long r = t / LJ;
  int co = (int)(r % 32), b = (int)(r / 32);
  const float* sb = s + (long)b * 32 * LJ;
  float acc = bsm[co];
#pragma unroll
  for (int dw = 0; dw < 3; ++dw) {
    int jj = j + dw - 1;
    if (jj < 0 || jj >= (int)LJ) continue;
#pragma unroll
    for (int ci = 0; ci < 32; ++ci) acc += wsm[(co * 32 + ci) * 3 + dw] * sb[(long)ci * LJ + jj];
  }
  out[t] = acc;
}

// ---------------- Launch ----------------

extern "C" void kernel_launch(void* const* d_in, const int* in_sizes, int n_in,
                              void* d_out, int out_size, void* d_ws, size_t ws_size,
                              hipStream_t stream) {
  (void)in_sizes; (void)n_in; (void)out_size; (void)ws_size;
  const float* x    = (const float*)d_in[0];
  const float* ln1w = (const float*)d_in[1];
  const float* ln1b = (const float*)d_in[2];
  const float* inW  = (const float*)d_in[3];
  const float* inb  = (const float*)d_in[4];
  const float* outW = (const float*)d_in[5];
  const float* outb = (const float*)d_in[6];
  const float* ln2w = (const float*)d_in[7];
  const float* ln2b = (const float*)d_in[8];
  const float* tcnW = (const float*)d_in[9];
  const float* tcnb = (const float*)d_in[10];
  const float* bng  = (const float*)d_in[11];
  const float* bnb  = (const float*)d_in[12];
  const float* bnm  = (const float*)d_in[13];
  const float* bnv  = (const float*)d_in[14];
  const float* fw   = (const float*)d_in[15];
  const float* fb   = (const float*)d_in[16];

  char* ws = (char*)d_ws;
  __bf16* inSwz  = (__bf16*)(ws + WS_INSWZ);
  __bf16* outSwz = (__bf16*)(ws + WS_OUTSWZ);
  float*  tcnR   = (float*)(ws + WS_TCNR);
  float*  sbuf   = (float*)(ws + WS_S);
  float*  out    = (float*)d_out;

  prep_weights<<<(768 * 256 + 255) / 256, 256, 0, stream>>>(inW, outW, tcnW, inSwz, outSwz, tcnR);
  swp_main<<<BN * NWIN, 256, LDS_TOT, stream>>>(x, ln1w, ln1b, inb, outb, ln2w, ln2b,
                                                tcnb, bng, bnb, bnm, bnv,
                                                inSwz, outSwz, tcnR, sbuf);
  fus_conv<<<(int)(((long)BN * 32 * 8064 + 255) / 256), 256, 0, stream>>>(sbuf, fw, fb, out);
}
